// MultiHeadSelfAttention_80204219286162
// MI455X (gfx1250) — matrix operand order repaired and checked by reading, not compiled, not run on hardware
//
#include <hip/hip_runtime.h>
#include <hip/hip_bf16.h>
#include <math.h>

// ---------------------------------------------------------------------------
// Multi-head causal self-attention for MI455X (gfx1250), bf16 WMMA pipeline.
//   B=4, T=2048, D=1024, H=16, HD=64
// ---------------------------------------------------------------------------

typedef __bf16 bf16_t;
typedef __attribute__((ext_vector_type(16))) __bf16 bf16x16;
typedef __attribute__((ext_vector_type(2)))  __bf16 bf16x2;
typedef __attribute__((ext_vector_type(8)))  float  f32x8;

#define WMMA_BF16(A, Bm, C) \
  __builtin_amdgcn_wmma_f32_16x16x32_bf16(false, (A), false, (Bm), (short)0, (C), false, false)

#define NB   4
#define NT   2048
#define ND   1024
#define NH   16
#define NHD  64
#define NBT  (NB * NT)          // 8192 token rows

// ----- fragment loaders ----------------------------------------------------
// A-matrix (16x32 bf16, row-major source): lane holds row m = lane%16.
// reg r covers K = kbase + 2*(r%4) + 16*(r/4) + 8*(lane/16) + {0,1}
__device__ __forceinline__ bf16x16 load_a_frag(const bf16_t* rowptr, int kbase, int hl) {
  bf16x16 a;
#pragma unroll
  for (int r = 0; r < 8; ++r) {
    int k = kbase + 2 * (r & 3) + 16 * (r >> 2) + 8 * hl;
    bf16x2 p = *(const bf16x2*)(rowptr + k);
    a[2 * r] = p.x; a[2 * r + 1] = p.y;
  }
  return a;
}

// B-matrix (32x16 bf16), source stored "N-major" (each N-row has contiguous K).
// lane holds column n = lane%16; reg r covers K = kbase + 2*(r%4) + 16*(r/4) + 8*(lane/16) + {0,1}
__device__ __forceinline__ bf16x16 load_b_frag(const bf16_t* rowptr, int kbase, int hl) {
  bf16x16 b;
#pragma unroll
  for (int r = 0; r < 8; ++r) {
    int k = kbase + 2 * (r & 3) + 16 * (r >> 2) + 8 * hl;
    bf16x2 p = *(const bf16x2*)(rowptr + k);
    b[2 * r] = p.x; b[2 * r + 1] = p.y;
  }
  return b;
}

// ----- conversion kernels --------------------------------------------------
__global__ __launch_bounds__(256) void cvt_f32_bf16(const float* __restrict__ x,
                                                    bf16_t* __restrict__ y, int n) {
  int i = blockIdx.x * 256 + threadIdx.x;
  if (i < n) y[i] = (bf16_t)x[i];
}

// wt[n*K + k] = (bf16) w[k*N + n]   (w is [K,N] row-major -> wt is [N,K])
__global__ __launch_bounds__(256) void cvt_transpose(const float* __restrict__ w,
                                                     bf16_t* __restrict__ wt,
                                                     int K, int N) {
  int i = blockIdx.x * 256 + threadIdx.x;
  if (i < K * N) {
    int n = i / K;
    int k = i - n * K;
    wt[(size_t)n * K + k] = (bf16_t)w[(size_t)k * N + n];
  }
}

// ----- QKV projection: [8192,1024] x [1024,3072] -> Q,K [B,H,T,64], V^T [B,H,64,T]
__global__ __launch_bounds__(256) void qkv_gemm(const bf16_t* __restrict__ xb,
                                                const bf16_t* __restrict__ wt,   // [3072,1024]
                                                bf16_t* __restrict__ qb,
                                                bf16_t* __restrict__ kbuf,
                                                bf16_t* __restrict__ vt) {
  const int lane = threadIdx.x & 31;
  const int hl = lane >> 4, l16 = lane & 15;
  // wave/tile ids are wave-uniform: pin them to SGPRs.
  const int wave = __builtin_amdgcn_readfirstlane((blockIdx.x * 256 + threadIdx.x) >> 5);
  const int NTILES = (3 * ND) / 64;                 // 48
  const int nt = wave % NTILES, mt = wave / NTILES; // mt < 512
  const int m0 = mt * 16, n0 = nt * 64;

  const bf16_t* arow = xb + (size_t)(m0 + l16) * ND;
  const bf16_t* b0r = wt + (size_t)(n0 + l16) * ND;
  const bf16_t* b1r = b0r + (size_t)16 * ND;
  const bf16_t* b2r = b0r + (size_t)32 * ND;
  const bf16_t* b3r = b0r + (size_t)48 * ND;
  f32x8 acc0 = {}, acc1 = {}, acc2 = {}, acc3 = {};

#pragma unroll 2
  for (int k0 = 0; k0 < ND; k0 += 32) {
    __builtin_prefetch(arow + k0 + 64, 0, 1);
    __builtin_prefetch(b0r + k0 + 64, 0, 1);
    // issue every load of this K-step before any WMMA consumes them
    bf16x16 a  = load_a_frag(arow, k0, hl);
    bf16x16 f0 = load_b_frag(b0r, k0, hl);
    bf16x16 f1 = load_b_frag(b1r, k0, hl);
    bf16x16 f2 = load_b_frag(b2r, k0, hl);
    bf16x16 f3 = load_b_frag(b3r, k0, hl);
    acc0 = WMMA_BF16(a, f0, acc0);
    acc1 = WMMA_BF16(a, f1, acc1);
    acc2 = WMMA_BF16(a, f2, acc2);
    acc3 = WMMA_BF16(a, f3, acc3);
  }

  // n0 is a multiple of 64 -> whole tile belongs to one segment and one head.
  const int seg = __builtin_amdgcn_readfirstlane(n0 >> 10);   // 0=Q 1=K 2=V
  const int h   = __builtin_amdgcn_readfirstlane((n0 & 1023) >> 6);
  bf16_t* base = (seg == 0) ? qb : ((seg == 1) ? kbuf : vt);
  const bool tr = (seg == 2);
  f32x8 accs[4] = {acc0, acc1, acc2, acc3};
#pragma unroll
  for (int a = 0; a < 4; ++a) {
#pragma unroll
    for (int p = 0; p < 8; ++p) {
      int row  = m0 + p + 8 * hl;       // global token row in [0,8192)
      int bidx = row >> 11;             // /2048
      int tloc = row & 2047;
      int hd   = a * 16 + l16;
      size_t bh_base = (size_t)(bidx * NH + h);
      size_t idx_rm  = (bh_base * NT + tloc) * NHD + hd;  // Q/K row-major
      size_t idx_tr  = (bh_base * NHD + hd) * NT + tloc;  // V transposed
      base[tr ? idx_tr : idx_rm] = (bf16_t)accs[a][p];
    }
  }
}

// ----- flash attention: one wave per 16-query tile -------------------------
__global__ __launch_bounds__(256) void attn_kernel(const bf16_t* __restrict__ qb,
                                                   const bf16_t* __restrict__ kb,
                                                   const bf16_t* __restrict__ vt,
                                                   bf16_t* __restrict__ ab) {
  __shared__ bf16_t pl[8][16 * 32];     // per-wave P tile (16 rows x 32 keys)
  const int wv = threadIdx.x >> 5;
  const int lane = threadIdx.x & 31;
  const int hl = lane >> 4, l16 = lane & 15;
  const int gw = __builtin_amdgcn_readfirstlane(blockIdx.x * 8 + wv);  // 0..8191
  const int qt = gw & 127;              // T/16 tiles
  const int bh = gw >> 7;               // 0..63
  const int q0 = qt * 16;

  const bf16_t* Q = qb + (size_t)bh * NT * NHD;
  const bf16_t* K = kb + (size_t)bh * NT * NHD;
  const bf16_t* V = vt + (size_t)bh * NHD * NT;

  const bf16_t* qrow = Q + (size_t)(q0 + l16) * NHD;
  bf16x16 aq0 = load_a_frag(qrow, 0, hl);
  bf16x16 aq1 = load_a_frag(qrow, 32, hl);

  f32x8 o0 = {}, o1 = {}, o2 = {}, o3 = {};
  float mrow[8], lrow[8];
#pragma unroll
  for (int p = 0; p < 8; ++p) { mrow[p] = -__builtin_inff(); lrow[p] = 0.f; }

  const float sc = 0.125f;              // 1/sqrt(64)

  for (int k0 = 0; k0 <= q0 + 15; k0 += 32) {   // k0 <= 2016, all loads in-bounds
    const bf16_t* kr0 = K + (size_t)(k0 + l16) * NHD;
    const bf16_t* kr1 = K + (size_t)(k0 + 16 + l16) * NHD;
    __builtin_prefetch(V + (size_t)l16 * NT + k0, 0, 1);
    // all four K fragments in flight before the score WMMAs
    bf16x16 kb00 = load_b_frag(kr0, 0, hl);
    bf16x16 kb01 = load_b_frag(kr0, 32, hl);
    bf16x16 kb10 = load_b_frag(kr1, 0, hl);
    bf16x16 kb11 = load_b_frag(kr1, 32, hl);
    f32x8 s0 = {}, s1 = {};
    s0 = WMMA_BF16(aq0, kb00, s0);
    s0 = WMMA_BF16(aq1, kb01, s0);
    s1 = WMMA_BF16(aq0, kb10, s1);
    s1 = WMMA_BF16(aq1, kb11, s1);

    const bool edge = (k0 + 31 > q0);
    float rmax[8];
#pragma unroll
    for (int p = 0; p < 8; ++p) {
      float v0 = s0[p] * sc, v1 = s1[p] * sc;
      int tok = q0 + p + 8 * hl;
      if (edge) {
        if (k0 + l16 > tok)      v0 = -__builtin_inff();
        if (k0 + 16 + l16 > tok) v1 = -__builtin_inff();
      }
      s0[p] = v0; s1[p] = v1;
      float r = fmaxf(v0, v1);
      r = fmaxf(r, __shfl_xor(r, 1, 32));
      r = fmaxf(r, __shfl_xor(r, 2, 32));
      r = fmaxf(r, __shfl_xor(r, 4, 32));
      r = fmaxf(r, __shfl_xor(r, 8, 32));
      rmax[p] = r;                       // identical across the 16-lane half
    }

#pragma unroll
    for (int p = 0; p < 8; ++p) {
      float mnew  = fmaxf(mrow[p], rmax[p]);
      float alpha = __expf(mrow[p] - mnew);
      float p0 = __expf(s0[p] - mnew);
      float p1 = __expf(s1[p] - mnew);
      float rs = p0 + p1;
      rs += __shfl_xor(rs, 1, 32);
      rs += __shfl_xor(rs, 2, 32);
      rs += __shfl_xor(rs, 4, 32);
      rs += __shfl_xor(rs, 8, 32);
      lrow[p] = lrow[p] * alpha + rs;
      mrow[p] = mnew;
      o0[p] *= alpha; o1[p] *= alpha; o2[p] *= alpha; o3[p] *= alpha;
      int row = p + 8 * hl;
      pl[wv][row * 32 + l16]      = (bf16_t)p0;
      pl[wv][row * 32 + 16 + l16] = (bf16_t)p1;
    }
    asm volatile("s_wait_dscnt 0" ::: "memory");   // intra-wave LDS RAW fence

    bf16x16 ap = load_a_frag(&pl[wv][l16 * 32], 0, hl);
    bf16x16 vb0 = load_b_frag(V + (size_t)(l16) * NT      + k0, 0, hl);
    bf16x16 vb1 = load_b_frag(V + (size_t)(16 + l16) * NT + k0, 0, hl);
    bf16x16 vb2 = load_b_frag(V + (size_t)(32 + l16) * NT + k0, 0, hl);
    bf16x16 vb3 = load_b_frag(V + (size_t)(48 + l16) * NT + k0, 0, hl);
    o0 = WMMA_BF16(ap, vb0, o0);
    o1 = WMMA_BF16(ap, vb1, o1);
    o2 = WMMA_BF16(ap, vb2, o2);
    o3 = WMMA_BF16(ap, vb3, o3);
  }

  const int bidx = bh >> 4, h = bh & 15;
#pragma unroll
  for (int p = 0; p < 8; ++p) {
    float inv = 1.0f / lrow[p];
    int row = p + 8 * hl;
    size_t orow = ((size_t)(bidx * NT + q0 + row)) * ND + h * NHD;
    ab[orow +      l16] = (bf16_t)(o0[p] * inv);
    ab[orow + 16 + l16] = (bf16_t)(o1[p] * inv);
    ab[orow + 32 + l16] = (bf16_t)(o2[p] * inv);
    ab[orow + 48 + l16] = (bf16_t)(o3[p] * inv);
  }
}

// ----- output projection: [8192,1024] x [1024,1024] -> f32 d_out -----------
__global__ __launch_bounds__(256) void outproj_gemm(const bf16_t* __restrict__ ab,
                                                    const bf16_t* __restrict__ wt, // [1024,1024] N-major
                                                    float* __restrict__ out) {
  const int lane = threadIdx.x & 31;
  const int hl = lane >> 4, l16 = lane & 15;
  const int wave = __builtin_amdgcn_readfirstlane((blockIdx.x * 256 + threadIdx.x) >> 5);
  const int NTILES = ND / 64;                       // 16
  const int nt = wave % NTILES, mt = wave / NTILES; // mt < 512
  const int m0 = mt * 16, n0 = nt * 64;

  const bf16_t* arow = ab + (size_t)(m0 + l16) * ND;
  const bf16_t* b0r = wt + (size_t)(n0 + l16) * ND;
  const bf16_t* b1r = b0r + (size_t)16 * ND;
  const bf16_t* b2r = b0r + (size_t)32 * ND;
  const bf16_t* b3r = b0r + (size_t)48 * ND;
  f32x8 acc0 = {}, acc1 = {}, acc2 = {}, acc3 = {};

#pragma unroll 2
  for (int k0 = 0; k0 < ND; k0 += 32) {
    __builtin_prefetch(arow + k0 + 64, 0, 1);
    __builtin_prefetch(b0r + k0 + 64, 0, 1);
    bf16x16 a  = load_a_frag(arow, k0, hl);
    bf16x16 f0 = load_b_frag(b0r, k0, hl);
    bf16x16 f1 = load_b_frag(b1r, k0, hl);
    bf16x16 f2 = load_b_frag(b2r, k0, hl);
    bf16x16 f3 = load_b_frag(b3r, k0, hl);
    acc0 = WMMA_BF16(a, f0, acc0);
    acc1 = WMMA_BF16(a, f1, acc1);
    acc2 = WMMA_BF16(a, f2, acc2);
    acc3 = WMMA_BF16(a, f3, acc3);
  }

  f32x8 accs[4] = {acc0, acc1, acc2, acc3};
#pragma unroll
  for (int a = 0; a < 4; ++a) {
#pragma unroll
    for (int p = 0; p < 8; ++p) {
      int row = m0 + p + 8 * hl;
      out[(size_t)row * ND + n0 + a * 16 + l16] = accs[a][p];
    }
  }
}

// ---------------------------------------------------------------------------
extern "C" void kernel_launch(void* const* d_in, const int* in_sizes, int n_in,
                              void* d_out, int out_size, void* d_ws, size_t ws_size,
                              hipStream_t stream) {
  const float* x     = (const float*)d_in[0];   // [4,2048,1024]
  const float* w_qkv = (const float*)d_in[1];   // [1024,3072]
  const float* w_out = (const float*)d_in[2];   // [1024,1024]
  float* out = (float*)d_out;                   // [4,2048,1024]

  char* ws = (char*)d_ws;
  const size_t MB = 1024ull * 1024ull;
  bf16_t* xb    = (bf16_t*)(ws);                 // 16 MB  [8192,1024]
  bf16_t* wqkvT = (bf16_t*)(ws + 16 * MB);       //  6 MB  [3072,1024]
  bf16_t* woutT = (bf16_t*)(ws + 22 * MB);       //  2 MB  [1024,1024]
  bf16_t* qb    = (bf16_t*)(ws + 24 * MB);       // 16 MB  [B,H,T,64]
  bf16_t* kb    = (bf16_t*)(ws + 40 * MB);       // 16 MB  [B,H,T,64]
  bf16_t* vt    = (bf16_t*)(ws + 56 * MB);       // 16 MB  [B,H,64,T]
  bf16_t* ab    = (bf16_t*)(ws + 72 * MB);       // 16 MB  [8192,1024]
  // total 88 MB of workspace

  // 1) precision conversion + weight transposes
  {
    int n = NBT * ND;
    cvt_f32_bf16<<<(n + 255) / 256, 256, 0, stream>>>(x, xb, n);
  }
  {
    int n = ND * 3 * ND;
    cvt_transpose<<<(n + 255) / 256, 256, 0, stream>>>(w_qkv, wqkvT, ND, 3 * ND);
  }
  {
    int n = ND * ND;
    cvt_transpose<<<(n + 255) / 256, 256, 0, stream>>>(w_out, woutT, ND, ND);
  }

  // 2) QKV projection: 512 M-tiles * 48 N-tiles = 24576 waves -> 3072 blocks
  qkv_gemm<<<3072, 256, 0, stream>>>(xb, wqkvT, qb, kb, vt);

  // 3) flash attention: 4*16*128 = 8192 query tiles -> 1024 blocks
  attn_kernel<<<1024, 256, 0, stream>>>(qb, kb, vt, ab);

  // 4) output projection: 512 * 16 = 8192 waves -> 1024 blocks
  outproj_gemm<<<1024, 256, 0, stream>>>(ab, woutT, out);
}
